// PaddedAttention_78898549227592
// MI455X (gfx1250) — matrix operand order repaired and checked by reading, not compiled, not run on hardware
//
#include <hip/hip_runtime.h>

// Problem constants
#define BN   8
#define SEQ  1024
#define CDIM 768
#define NH   12
#define HD   64
#define QKVC (3 * CDIM)          // 2304
static __device__ __forceinline__ float attn_scale() { return 0.125f; } // 1/sqrt(64)

typedef __bf16 bf16;
typedef __attribute__((ext_vector_type(16))) __bf16 v16bf;
typedef __attribute__((ext_vector_type(8)))  float  v8f;

union FragU { v16bf v; uint4 q[2]; };

// ---------------------------------------------------------------------------
// WMMA fragment helpers (wave32 layouts per CDNA5 ISA 7.12.2)
// ---------------------------------------------------------------------------

// A-matrix 16x32 bf16, row-major source (K contiguous), stride lda (elements).
// lanes 0-15: row=lane, halves 0..7 = K0..7, 8..15 = K16..23
// lanes16-31: row=lane-16, halves 0..7 = K8..15, 8..15 = K24..31
__device__ __forceinline__ v16bf load_frag_a(const bf16* A, int lda, int m0, int k0, int lane) {
    int half = lane >> 4;
    int r    = lane & 15;
    const bf16* p = A + (size_t)(m0 + r) * lda + k0 + half * 8;
    FragU u;
    u.q[0] = *(const uint4*)(p);
    u.q[1] = *(const uint4*)(p + 16);
    return u.v;
}

// B-matrix 32x16 bf16 from Bt[n][k] (K contiguous), stride ldb (elements).
// lanes 0-15: col=lane, halves 0..7 = K0..7, 8..15 = K16..23 ; lanes 16-31: col=lane-16, K8..15, K24..31
__device__ __forceinline__ v16bf load_frag_b(const bf16* Bt, int ldb, int n0, int k0, int lane) {
    int half = lane >> 4;
    int c    = lane & 15;
    const bf16* p = Bt + (size_t)(n0 + c) * ldb + k0 + half * 8;
    FragU u;
    u.q[0] = *(const uint4*)(p);
    u.q[1] = *(const uint4*)(p + 16);
    return u.v;
}

__device__ __forceinline__ v8f wmma_bf16(v16bf a, v16bf b, v8f c) {
    return __builtin_amdgcn_wmma_f32_16x16x32_bf16(false, a, false, b, (short)0, c, false, false);
}

// cross-lane reductions over the 16-lane half-wave (C-layout row reductions)
__device__ __forceinline__ float hmax16(float v) {
    v = fmaxf(v, __shfl_xor(v, 1, 32));
    v = fmaxf(v, __shfl_xor(v, 2, 32));
    v = fmaxf(v, __shfl_xor(v, 4, 32));
    v = fmaxf(v, __shfl_xor(v, 8, 32));
    return v;
}
__device__ __forceinline__ float hsum16(float v) {
    v += __shfl_xor(v, 1, 32);
    v += __shfl_xor(v, 2, 32);
    v += __shfl_xor(v, 4, 32);
    v += __shfl_xor(v, 8, 32);
    return v;
}

// ---------------------------------------------------------------------------
// Prep kernels: fp32 -> bf16 staging
// ---------------------------------------------------------------------------
__global__ void cvt_f32_bf16(const float* __restrict__ src, bf16* __restrict__ dst, int n) {
    int i = blockIdx.x * 256 + threadIdx.x;
    if (i < n) dst[i] = (bf16)src[i];
}

// Wt[j*rows + k] = W[k*cols + j]  (rows = K dim, cols = output dim)
__global__ void transpose_w_bf16(const float* __restrict__ W, bf16* __restrict__ Wt,
                                 int rows, int cols) {
    int t = blockIdx.x * 256 + threadIdx.x;
    if (t >= rows * cols) return;
    int j = t / rows;
    int k = t - j * rows;
    Wt[t] = (bf16)W[(size_t)k * cols + j];
}

// ---------------------------------------------------------------------------
// QKV GEMM: [8192,768] @ [768,2304] -> scatter to Q[b,h,n,d], K[b,h,n,d], Vt[b,h,d,n]
// block = 128 threads (4 waves), wave tile = 32x64, block tile = 128x64
// ---------------------------------------------------------------------------
__global__ void gemm_qkv_kernel(const bf16* __restrict__ Xb, const bf16* __restrict__ Wt,
                                const float* __restrict__ bias,
                                bf16* __restrict__ Qb, bf16* __restrict__ Kb,
                                bf16* __restrict__ Vt) {
    int lane = threadIdx.x & 31;
    int wave = threadIdx.x >> 5;
    int m0 = blockIdx.x * 128 + wave * 32;
    int n0 = blockIdx.y * 64;

    v8f acc[2][4];
#pragma unroll
    for (int r = 0; r < 2; ++r)
#pragma unroll
        for (int j = 0; j < 4; ++j) acc[r][j] = (v8f){};

    for (int k0 = 0; k0 < CDIM; k0 += 32) {
        v16bf a0 = load_frag_a(Xb, CDIM, m0,      k0, lane);
        v16bf a1 = load_frag_a(Xb, CDIM, m0 + 16, k0, lane);
#pragma unroll
        for (int j = 0; j < 4; ++j) {
            v16bf bf = load_frag_b(Wt, CDIM, n0 + j * 16, k0, lane);
            acc[0][j] = wmma_bf16(a0, bf, acc[0][j]);
            acc[1][j] = wmma_bf16(a1, bf, acc[1][j]);
        }
    }

    int half = lane >> 4;
    int c    = lane & 15;
#pragma unroll
    for (int r = 0; r < 2; ++r) {
#pragma unroll
        for (int j = 0; j < 4; ++j) {
            int col  = n0 + j * 16 + c;
            float bv = bias[col];
            int three = col / CDIM;
            int rem   = col % CDIM;
            int hh    = rem >> 6;      // head
            int d     = rem & 63;      // dim within head
#pragma unroll
            for (int i = 0; i < 8; ++i) {
                int row = m0 + r * 16 + half * 8 + i;
                int b = row >> 10;
                int n = row & 1023;
                float v = acc[r][j][i] + bv;
                size_t bh = (size_t)b * NH + hh;
                if (three == 0)      Qb[(bh * SEQ + n) * HD + d] = (bf16)v;
                else if (three == 1) Kb[(bh * SEQ + n) * HD + d] = (bf16)v;
                else                 Vt[(bh * HD + d) * SEQ + n] = (bf16)v;
            }
        }
    }
}

// ---------------------------------------------------------------------------
// Flash attention: one wave per 16 query rows, online softmax over 32-key tiles
// grid = (SEQ/64, B*H), block = 128 (4 waves)
// ---------------------------------------------------------------------------
__global__ void attn_kernel(const bf16* __restrict__ Qb, const bf16* __restrict__ Kb,
                            const bf16* __restrict__ Vt, const float* __restrict__ mask,
                            bf16* __restrict__ Ctx) {
    __shared__ __align__(16) bf16 plds[4][16 * 32];

    int lane = threadIdx.x & 31;
    int wave = threadIdx.x >> 5;
    int bh = blockIdx.y;
    int b  = bh / NH;
    int h  = bh % NH;
    int q0 = blockIdx.x * 64 + wave * 16;

    const bf16* Qp = Qb + (size_t)bh * SEQ * HD;
    const bf16* Kp = Kb + (size_t)bh * SEQ * HD;
    const bf16* Vp = Vt + (size_t)bh * HD * SEQ;
    const float* mrow = mask + (size_t)b * SEQ;

    int half = lane >> 4;
    int c    = lane & 15;

    v16bf qa0 = load_frag_a(Qp, HD, q0, 0,  lane);
    v16bf qa1 = load_frag_a(Qp, HD, q0, 32, lane);

    float run_max[8], run_sum[8];
    v8f oacc[4];
#pragma unroll
    for (int i = 0; i < 8; ++i) { run_max[i] = -1.0e30f; run_sum[i] = 0.0f; }
#pragma unroll
    for (int dj = 0; dj < 4; ++dj) oacc[dj] = (v8f){};

    bf16* pl = plds[wave];

    for (int kt = 0; kt < SEQ; kt += 32) {
        // scores: two 16x16 tiles (keys kt..kt+15, kt+16..kt+31)
        v8f s0 = (v8f){};
        v8f s1 = (v8f){};
        s0 = wmma_bf16(qa0, load_frag_b(Kp, HD, kt,      0,  lane), s0);
        s0 = wmma_bf16(qa1, load_frag_b(Kp, HD, kt,      32, lane), s0);
        s1 = wmma_bf16(qa0, load_frag_b(Kp, HD, kt + 16, 0,  lane), s1);
        s1 = wmma_bf16(qa1, load_frag_b(Kp, HD, kt + 16, 32, lane), s1);

        float m0v = mrow[kt + c];
        float m1v = mrow[kt + 16 + c];

#pragma unroll
        for (int i = 0; i < 8; ++i) {
            float a = s0[i] * attn_scale() + m0v;
            float e = s1[i] * attn_scale() + m1v;
            float mx = hmax16(fmaxf(a, e));
            float nm = fmaxf(run_max[i], mx);
            float corr = __expf(run_max[i] - nm);
            float p0 = __expf(a - nm);
            float p1 = __expf(e - nm);
            float rs = hsum16(p0 + p1);
            run_sum[i] = run_sum[i] * corr + rs;
            run_max[i] = nm;
#pragma unroll
            for (int dj = 0; dj < 4; ++dj) oacc[dj][i] *= corr;
            // stage P tile (16x32) in LDS for A-fragment reload
            pl[(i + half * 8) * 32 + c]      = (bf16)p0;
            pl[(i + half * 8) * 32 + c + 16] = (bf16)p1;
        }

        asm volatile("s_wait_dscnt 0x0" ::: "memory");
        v16bf pa = load_frag_a(pl, 32, 0, 0, lane);

#pragma unroll
        for (int dj = 0; dj < 4; ++dj)
            oacc[dj] = wmma_bf16(pa, load_frag_b(Vp, SEQ, dj * 16, kt, lane), oacc[dj]);
    }

    // normalize and scatter to context [B, N, H*D] (bf16, A-layout for proj GEMM)
#pragma unroll
    for (int i = 0; i < 8; ++i) {
        int row = q0 + half * 8 + i;
        float inv = 1.0f / run_sum[i];
#pragma unroll
        for (int dj = 0; dj < 4; ++dj) {
            int cc = h * HD + dj * 16 + c;
            Ctx[((size_t)b * SEQ + row) * CDIM + cc] = (bf16)(oacc[dj][i] * inv);
        }
    }
}

// ---------------------------------------------------------------------------
// Output projection: [8192,768] @ [768,768] + bias -> fp32 d_out
// ---------------------------------------------------------------------------
__global__ void gemm_proj_kernel(const bf16* __restrict__ Ctx, const bf16* __restrict__ Wt,
                                 const float* __restrict__ bias, float* __restrict__ out) {
    int lane = threadIdx.x & 31;
    int wave = threadIdx.x >> 5;
    int m0 = blockIdx.x * 128 + wave * 32;
    int n0 = blockIdx.y * 64;

    v8f acc[2][4];
#pragma unroll
    for (int r = 0; r < 2; ++r)
#pragma unroll
        for (int j = 0; j < 4; ++j) acc[r][j] = (v8f){};

    for (int k0 = 0; k0 < CDIM; k0 += 32) {
        v16bf a0 = load_frag_a(Ctx, CDIM, m0,      k0, lane);
        v16bf a1 = load_frag_a(Ctx, CDIM, m0 + 16, k0, lane);
#pragma unroll
        for (int j = 0; j < 4; ++j) {
            v16bf bf = load_frag_b(Wt, CDIM, n0 + j * 16, k0, lane);
            acc[0][j] = wmma_bf16(a0, bf, acc[0][j]);
            acc[1][j] = wmma_bf16(a1, bf, acc[1][j]);
        }
    }

    int half = lane >> 4;
    int c    = lane & 15;
#pragma unroll
    for (int r = 0; r < 2; ++r) {
#pragma unroll
        for (int j = 0; j < 4; ++j) {
            int col = n0 + j * 16 + c;
            float bv = bias[col];
#pragma unroll
            for (int i = 0; i < 8; ++i) {
                int row = m0 + r * 16 + half * 8 + i;
                out[(size_t)row * CDIM + col] = acc[r][j][i] + bv;
            }
        }
    }
}

// ---------------------------------------------------------------------------
// Host launcher
// ---------------------------------------------------------------------------
extern "C" void kernel_launch(void* const* d_in, const int* in_sizes, int n_in,
                              void* d_out, int out_size, void* d_ws, size_t ws_size,
                              hipStream_t stream) {
    const float* x     = (const float*)d_in[0];
    const float* mask  = (const float*)d_in[1];
    const float* Wqkv  = (const float*)d_in[2];
    const float* bqkv  = (const float*)d_in[3];
    const float* Wproj = (const float*)d_in[4];
    const float* bproj = (const float*)d_in[5];
    float* out = (float*)d_out;

    char* ws = (char*)d_ws;
    const size_t TOK   = (size_t)BN * SEQ;           // 8192
    const size_t XEL   = TOK * CDIM;                 // 6,291,456
    const size_t QKVEL = (size_t)BN * NH * SEQ * HD; // 6,291,456 each

    bf16* Xb      = (bf16*)(ws);
    bf16* Qb      = (bf16*)(ws + 12582912);
    bf16* Kb      = (bf16*)(ws + 25165824);
    bf16* Vt      = (bf16*)(ws + 37748736);
    bf16* Ctx     = (bf16*)(ws + 50331648);
    bf16* Wqkv_t  = (bf16*)(ws + 62914560);          // 2304*768 bf16
    bf16* Wproj_t = (bf16*)(ws + 66453504);          // 768*768 bf16
    (void)QKVEL; (void)in_sizes; (void)n_in; (void)out_size; (void)ws_size;

    // 1) stage x as bf16
    cvt_f32_bf16<<<(int)((XEL + 255) / 256), 256, 0, stream>>>(x, Xb, (int)XEL);
    // 2) transposed bf16 weights (K contiguous -> direct B fragments)
    transpose_w_bf16<<<(CDIM * QKVC + 255) / 256, 256, 0, stream>>>(Wqkv, Wqkv_t, CDIM, QKVC);
    transpose_w_bf16<<<(CDIM * CDIM + 255) / 256, 256, 0, stream>>>(Wproj, Wproj_t, CDIM, CDIM);
    // 3) QKV projection -> Q / K / V^T staging
    gemm_qkv_kernel<<<dim3(TOK / 128, QKVC / 64), 128, 0, stream>>>(Xb, Wqkv_t, bqkv, Qb, Kb, Vt);
    // 4) flash attention per head -> context bf16
    attn_kernel<<<dim3(SEQ / 64, BN * NH), 128, 0, stream>>>(Qb, Kb, Vt, mask, Ctx);
    // 5) output projection -> fp32 d_out
    gemm_proj_kernel<<<dim3(TOK / 128, CDIM / 64), 128, 0, stream>>>(Ctx, Wproj_t, bproj, out);
}
